// MoEGAT_2559800509232
// MI455X (gfx1250) — matrix operand order repaired and checked by reading, not compiled, not run on hardware
//
#include <hip/hip_runtime.h>
#include <math.h>

// ---------------------------------------------------------------------------
// MoEGAT on MI455X (gfx1250, wave32, WMMA).
// All conv1d / einsum contractions run through one generalized batched GEMM
// that stages fp32 tiles into LDS as bf16 and issues v_wmma_f32_16x16x32_bf16
// with fp32 accumulators. Two tile configs: 128x128 block / 32x64 wave
// (8 WMMAs per K-step) for the large convs, 64x64 / 16x32 for small-M GEMMs.
// Full tiles take a vectorized float2 staging path (no per-element guards).
// ---------------------------------------------------------------------------

typedef __attribute__((ext_vector_type(16))) __bf16 v16bf;
typedef __attribute__((ext_vector_type(8)))  float  v8f;

union FragV { unsigned int u[8]; v16bf v; };

__device__ __forceinline__ unsigned short f2bfbits(float f) {
  unsigned u = __builtin_bit_cast(unsigned, f);
  unsigned r = u + 0x7FFFu + ((u >> 16) & 1u);   // round-to-nearest-even
  return (unsigned short)(r >> 16);
}
__device__ __forceinline__ unsigned pack_bf2(float x, float y) {
  return (unsigned)f2bfbits(x) | ((unsigned)f2bfbits(y) << 16);
}

// ---------------------------------------------------------------------------
// Generalized batched GEMM:  C[b] = act( op(A[b]) * op(B[b]) + bias + R[b] )
// A is M x K (row-major, lda; transA: A[m,k]=A[k*lda+m])
// B is K x N (row-major, ldb; transB: B[k,n]=B[n*ldb+k])
// Block tile BM x BN, wave tile TM x TN, BK=32, 8 wave32s per block.
// ---------------------------------------------------------------------------
template<int BM, int BN, int TM, int TN>
__global__ __launch_bounds__(256) void k_gemm(
    const float* __restrict__ A,  long long sA, int lda, int tA,
    const float* __restrict__ Bm, long long sB, int ldb, int tB,
    float* __restrict__ Cm,       long long sC, int ldc,
    const float* __restrict__ bias,
    const float* __restrict__ R,  long long sR, int ldr,
    int M, int N, int K, int relu)
{
  constexpr int RM = TM / 16;
  constexpr int RN = TN / 16;
  constexpr int WAVES_M = BM / TM;

  const int bz = blockIdx.z;
  A  += (long long)bz * sA;
  Bm += (long long)bz * sB;
  Cm += (long long)bz * sC;
  if (R) R += (long long)bz * sR;

  const int m0 = blockIdx.y * BM;
  const int n0 = blockIdx.x * BN;

  __shared__ unsigned short As[BM * 32];   // [row][k]  bf16
  __shared__ unsigned short Bs[BN * 32];   // [col][k]  bf16
  unsigned int* As32 = (unsigned int*)As;
  unsigned int* Bs32 = (unsigned int*)Bs;

  const int tid   = threadIdx.x;
  const int lane  = tid & 31;
  const int wave  = tid >> 5;
  const int wm    = wave % WAVES_M;
  const int wn    = wave / WAVES_M;
  const int lrow  = lane & 15;
  const int lhalf = lane >> 4;

  v8f acc[RM][RN];
  {
    v8f z = {};
#pragma unroll
    for (int r = 0; r < RM; ++r)
#pragma unroll
      for (int c = 0; c < RN; ++c) acc[r][c] = z;
  }

  for (int k0 = 0; k0 < K; k0 += 32) {
    const bool fullA = (m0 + BM <= M) && (k0 + 32 <= K);
    const bool fullB = (n0 + BN <= N) && (k0 + 32 <= K);

    // ---- stage A tile (BM x 32) as bf16, layout As[row][k] ----
    if (fullA) {
      if (!tA) {
        const float* Ap = A + (long long)m0 * lda + k0;
#pragma unroll
        for (int j = 0; j < (BM * 16) / 256; ++j) {
          int i = tid + j * 256;
          int ml = i >> 4, kp = i & 15;
          float2 v = *(const float2*)(Ap + (long long)ml * lda + 2 * kp);
          As32[ml * 16 + kp] = pack_bf2(v.x, v.y);
        }
        if (k0 + 32 < K)   // prefetch next K tile (global_prefetch_b8)
          __builtin_prefetch(Ap + (long long)(tid & (BM - 1)) * lda + 32, 0, 0);
      } else {
        const float* Ap = A + (long long)k0 * lda + m0;
#pragma unroll
        for (int j = 0; j < (BM * 32) / 256; ++j) {
          int i = tid + j * 256;
          int ml = i % BM, kl = i / BM;        // contiguous along m in memory
          As[ml * 32 + kl] = f2bfbits(Ap[(long long)kl * lda + ml]);
        }
      }
    } else {
      for (int i = tid; i < BM * 32; i += 256) {
        int ml = i >> 5, kl = i & 31;
        int gm = m0 + ml, gk = k0 + kl;
        float v = 0.f;
        if (gm < M && gk < K)
          v = tA ? A[(long long)gk * lda + gm] : A[(long long)gm * lda + gk];
        As[ml * 32 + kl] = f2bfbits(v);
      }
    }

    // ---- stage B tile as bf16, layout Bs[col][k] ----
    if (fullB) {
      if (tB) {
        const float* Bp = Bm + (long long)n0 * ldb + k0;
#pragma unroll
        for (int j = 0; j < (BN * 16) / 256; ++j) {
          int i = tid + j * 256;
          int col = i >> 4, kp = i & 15;
          float2 v = *(const float2*)(Bp + (long long)col * ldb + 2 * kp);
          Bs32[col * 16 + kp] = pack_bf2(v.x, v.y);
        }
      } else {
        const float* Bp = Bm + (long long)k0 * ldb + n0;
#pragma unroll
        for (int j = 0; j < (BN * 16) / 256; ++j) {
          int i = tid + j * 256;
          int kl = i / (BN / 2), cp = i % (BN / 2);
          int col = 2 * cp;
          float2 v = *(const float2*)(Bp + (long long)kl * ldb + col);
          Bs[col * 32 + kl]       = f2bfbits(v.x);
          Bs[(col + 1) * 32 + kl] = f2bfbits(v.y);
        }
      }
    } else {
      for (int i = tid; i < BN * 32; i += 256) {
        int col, kl;
        if (tB) { col = i >> 5; kl = i & 31; }
        else    { kl = i / BN;  col = i % BN; }
        int gn = n0 + col, gk = k0 + kl;
        float v = 0.f;
        if (gn < N && gk < K)
          v = tB ? Bm[(long long)gn * ldb + gk] : Bm[(long long)gk * ldb + gn];
        Bs[col * 32 + kl] = f2bfbits(v);
      }
    }
    __syncthreads();

    // ---- fragments per ISA 7.12.2 wave32 bf16 layout ----
    FragV af[RM], bf[RN];
#pragma unroll
    for (int r = 0; r < RM; ++r) {
      int ar = wm * TM + r * 16 + lrow;
      int kb = lhalf * 4;
#pragma unroll
      for (int j = 0; j < 4; ++j) {
        af[r].u[j]     = As32[ar * 16 + kb + j];       // K = kbase+2j, +1
        af[r].u[j + 4] = As32[ar * 16 + 8 + kb + j];   // K = 16+kbase+2j, +1
      }
    }
#pragma unroll
    for (int c = 0; c < RN; ++c) {
      int bc = wn * TN + c * 16 + lrow;
      int kb2 = lhalf * 4;
#pragma unroll
      for (int j = 0; j < 4; ++j) {
        bf[c].u[j]     = Bs32[bc * 16 + kb2 + j];
        bf[c].u[j + 4] = Bs32[bc * 16 + 8 + kb2 + j];
      }
    }
#pragma unroll
    for (int r = 0; r < RM; ++r)
#pragma unroll
      for (int c = 0; c < RN; ++c)
        acc[r][c] = __builtin_amdgcn_wmma_f32_16x16x32_bf16(
            false, af[r].v, false, bf[c].v, (short)0, acc[r][c], false, false);

    __syncthreads();
  }

  // ---- epilogue: lane half carries rows +0..7 / +8..15 of each 16x16 tile --
#pragma unroll
  for (int r = 0; r < RM; ++r) {
    int mb = m0 + wm * TM + r * 16 + lhalf * 8;
#pragma unroll
    for (int c = 0; c < RN; ++c) {
      int gn = n0 + wn * TN + c * 16 + lrow;
      if (gn >= N) continue;
#pragma unroll
      for (int i = 0; i < 8; ++i) {
        int gm = mb + i;
        if (gm >= M) continue;
        float v = acc[r][c][i];
        if (bias) v += bias[gm];
        if (R)    v += R[(long long)gm * ldr + gn];
        if (relu) v = fmaxf(v, 0.f);
        Cm[(long long)gm * ldc + gn] = v;
      }
    }
  }
}

// ---------------------------------------------------------------------------
// Training-mode BatchNorm over (batch, spatial), biased variance.
// One block per channel.
// ---------------------------------------------------------------------------
__global__ __launch_bounds__(256) void k_bnorm(
    const float* __restrict__ X, float* __restrict__ Y,
    const float* __restrict__ g, const float* __restrict__ b,
    int Bn, int C, int N, int relu, float eps)
{
  int c = blockIdx.x;
  __shared__ float s1[256], s2[256];
  float sum = 0.f, sq = 0.f;
  for (int bb = 0; bb < Bn; ++bb) {
    const float* p = X + ((long long)bb * C + c) * N;
    for (int n = threadIdx.x; n < N; n += 256) {
      float v = p[n]; sum += v; sq += v * v;
    }
  }
  s1[threadIdx.x] = sum; s2[threadIdx.x] = sq;
  __syncthreads();
  for (int s = 128; s > 0; s >>= 1) {
    if (threadIdx.x < s) { s1[threadIdx.x] += s1[threadIdx.x + s];
                           s2[threadIdx.x] += s2[threadIdx.x + s]; }
    __syncthreads();
  }
  float cnt  = (float)Bn * (float)N;
  float mean = s1[0] / cnt;
  float var  = s2[0] / cnt - mean * mean;
  float sc = g[c] * rsqrtf(var + eps);
  float sh = b[c] - mean * sc;
  for (int bb = 0; bb < Bn; ++bb) {
    const float* p = X + ((long long)bb * C + c) * N;
    float*       q = Y + ((long long)bb * C + c) * N;
    for (int n = threadIdx.x; n < N; n += 256) {
      float v = p[n] * sc + sh;
      if (relu) v = fmaxf(v, 0.f);
      q[n] = v;
    }
  }
}

// InstanceNorm over last dim, one block per (b,c) row, no affine.
__global__ __launch_bounds__(256) void k_instnorm(
    const float* __restrict__ X, float* __restrict__ Y, int N, float eps)
{
  long long r = blockIdx.x;
  const float* p = X + r * N;
  float*       q = Y + r * N;
  __shared__ float s1[256], s2[256];
  float sum = 0.f, sq = 0.f;
  for (int n = threadIdx.x; n < N; n += 256) { float v = p[n]; sum += v; sq += v * v; }
  s1[threadIdx.x] = sum; s2[threadIdx.x] = sq;
  __syncthreads();
  for (int s = 128; s > 0; s >>= 1) {
    if (threadIdx.x < s) { s1[threadIdx.x] += s1[threadIdx.x + s];
                           s2[threadIdx.x] += s2[threadIdx.x + s]; }
    __syncthreads();
  }
  float mean = s1[0] / N;
  float var  = s2[0] / N - mean * mean;
  float rs = rsqrtf(var + eps);
  for (int n = threadIdx.x; n < N; n += 256) q[n] = (p[n] - mean) * rs;
}

// Row softmax with optional kv mask and pre-softmax scale (all rows of a slab
// share one mask row). Masked logits become -1e6 before scaling (as reference).
__global__ __launch_bounds__(256) void k_softmax(
    const float* __restrict__ X, float* __restrict__ Y, int Mlen, float scale,
    const unsigned char* __restrict__ mask)
{
  long long r = blockIdx.x;
  const float* p = X + r * (long long)Mlen;
  float*       q = Y + r * (long long)Mlen;
  __shared__ float s[256];
  float mx = -3.4e38f;
  for (int m = threadIdx.x; m < Mlen; m += 256) {
    float v = p[m];
    if (mask && !mask[m]) v = -1.0e6f;
    v *= scale;
    mx = fmaxf(mx, v);
  }
  s[threadIdx.x] = mx; __syncthreads();
  for (int st = 128; st > 0; st >>= 1) {
    if (threadIdx.x < st) s[threadIdx.x] = fmaxf(s[threadIdx.x], s[threadIdx.x + st]);
    __syncthreads();
  }
  float rowmax = s[0];
  __syncthreads();
  float sum = 0.f;
  for (int m = threadIdx.x; m < Mlen; m += 256) {
    float v = p[m];
    if (mask && !mask[m]) v = -1.0e6f;
    float e = __expf(v * scale - rowmax);
    q[m] = e;
    sum += e;
  }
  s[threadIdx.x] = sum; __syncthreads();
  for (int st = 128; st > 0; st >>= 1) {
    if (threadIdx.x < st) s[threadIdx.x] += s[threadIdx.x + st];
    __syncthreads();
  }
  float inv = 1.f / s[0];
  for (int m = threadIdx.x; m < Mlen; m += 256) q[m] *= inv;
}

// Y[b, 0:2C, n] = concat(X1[b,:,n], X2[b,:,n]) along channels.
__global__ void k_concat2(const float* __restrict__ X1, const float* __restrict__ X2,
                          float* __restrict__ Y, int Bn, int C, int N)
{
  long long total = (long long)Bn * 2 * C * N;
  for (long long i = (long long)blockIdx.x * blockDim.x + threadIdx.x; i < total;
       i += (long long)gridDim.x * blockDim.x) {
    long long n  = i % N;
    long long c2 = (i / N) % (2 * C);
    long long b  = i / ((long long)N * 2 * C);
    float v = (c2 < C) ? X1[(b * C + c2) * N + n]
                       : X2[(b * C + (c2 - C)) * N + n];
    Y[i] = v;
  }
}

// Y[i] = X[i] * w[i / perBatch]  (per-batch scalar gate)
__global__ void k_scale_batch(const float* __restrict__ X, const float* __restrict__ w,
                              float* __restrict__ Y, long long perBatch, long long total)
{
  for (long long i = (long long)blockIdx.x * blockDim.x + threadIdx.x; i < total;
       i += (long long)gridDim.x * blockDim.x)
    Y[i] = X[i] * w[i / perBatch];
}

// Row means (adaptive avg pool): one block per row.
__global__ __launch_bounds__(256) void k_mean_rows(
    const float* __restrict__ X, float* __restrict__ Y, int len)
{
  long long r = blockIdx.x;
  const float* p = X + r * (long long)len;
  __shared__ float s[256];
  float sum = 0.f;
  for (int i = threadIdx.x; i < len; i += 256) sum += p[i];
  s[threadIdx.x] = sum; __syncthreads();
  for (int st = 128; st > 0; st >>= 1) {
    if (threadIdx.x < st) s[threadIdx.x] += s[threadIdx.x + st];
    __syncthreads();
  }
  if (threadIdx.x == 0) Y[r] = s[0] / (float)len;
}

// Rectify gate: w[b] = m2( relu(bnorm(m1(y[b]))) ), spatial size 1 so the BN
// stats are over the batch dimension only. Single block, 256 threads (== C).
__global__ __launch_bounds__(256) void k_rect_gate(
    const float* __restrict__ y,                        // [B, C] pooled means
    const float* __restrict__ m1w, const float* __restrict__ m1b,
    const float* __restrict__ gg,  const float* __restrict__ gb,
    const float* __restrict__ m2w, const float* __restrict__ m2b,
    float* __restrict__ wOut, int Bn, int C)
{
  __shared__ float t1[8 * 256];
  __shared__ float red[256];
  int c = threadIdx.x;
  for (int b = 0; b < Bn; ++b) {
    float acc = m1b[c];
    const float* yr = y + (long long)b * C;
    const float* wr = m1w + (long long)c * C;
    for (int k = 0; k < C; ++k) acc += wr[k] * yr[k];
    t1[b * C + c] = acc;
  }
  __syncthreads();
  float mean = 0.f;
  for (int b = 0; b < Bn; ++b) mean += t1[b * C + c];
  mean /= (float)Bn;
  float var = 0.f;
  for (int b = 0; b < Bn; ++b) { float d = t1[b * C + c] - mean; var += d * d; }
  var /= (float)Bn;
  float sc = gg[c] * rsqrtf(var + 1e-5f);
  float sh = gb[c] - mean * sc;
  for (int b = 0; b < Bn; ++b)
    t1[b * C + c] = fmaxf(t1[b * C + c] * sc + sh, 0.f);
  __syncthreads();
  for (int b = 0; b < Bn; ++b) {
    red[c] = m2w[c] * t1[b * C + c];
    __syncthreads();
    for (int st = 128; st > 0; st >>= 1) {
      if (c < st) red[c] += red[c + st];
      __syncthreads();
    }
    if (c == 0) wOut[b] = red[0] + m2b[0];
    __syncthreads();
  }
}

__global__ void k_zero(float* __restrict__ p, long long n)
{
  for (long long i = (long long)blockIdx.x * blockDim.x + threadIdx.x; i < n;
       i += (long long)gridDim.x * blockDim.x) p[i] = 0.f;
}

// MoE gate: top-2 of E=8 logits per token, softmax the pair, scatter weights,
// accumulate usage. logits layout: [B, E, P].
__global__ void k_moe_gate(const float* __restrict__ logits,
                           float* __restrict__ weight, float* __restrict__ usage,
                           int T, int E, int P)
{
  int t = blockIdx.x * blockDim.x + threadIdx.x;
  if (t >= T) return;
  int b = t / P, p = t % P;
  const float* lp = logits + (long long)b * E * P + p;
  float v1 = -3.4e38f, v2 = -3.4e38f;
  int i1 = 0, i2 = 0;
  for (int e = 0; e < E; ++e) {
    float v = lp[(long long)e * P];
    if (v > v1)      { v2 = v1; i2 = i1; v1 = v; i1 = e; }
    else if (v > v2) { v2 = v;  i2 = e; }
  }
  float e2 = __expf(v2 - v1);
  float inv = 1.f / (1.f + e2);
  float w1 = inv, w2 = e2 * inv;
  float* wp = weight + (long long)t * E;
  for (int e = 0; e < E; ++e) wp[e] = 0.f;
  wp[i1] = w1; wp[i2] = w2;
  atomicAdd(&usage[i1], w1);
  atomicAdd(&usage[i2], w2);
}

// Fused dense-all-experts MLP + top-2 weighting + residual.
// One block per token (t = b*P + p), 128 threads (E*Hh = 8*16).
__global__ __launch_bounds__(128) void k_moe_expert(
    const float* __restrict__ sp,      // [B, C, P], flat input & residual
    const float* __restrict__ weight,  // [T, E]
    const float* __restrict__ w1, const float* __restrict__ b1,  // [E,16,C],[E,16]
    const float* __restrict__ w2, const float* __restrict__ b2,  // [E,C,16],[E,C]
    float* __restrict__ out, int Bn, int C, int P, int E)
{
  int t = blockIdx.x;
  int b = t / P, p = t % P;
  __shared__ float xs[256];
  __shared__ float hs[8 * 16];
  const float* xp = sp + (long long)b * C * P + p;
  for (int c = threadIdx.x; c < C; c += 128) xs[c] = xp[(long long)c * P];
  __syncthreads();
  {
    int e = threadIdx.x >> 4, j = threadIdx.x & 15;
    const float* wrow = w1 + ((long long)e * 16 + j) * C;
    float acc = b1[e * 16 + j];
    for (int c = 0; c < C; ++c) acc += wrow[c] * xs[c];
    hs[e * 16 + j] = fmaxf(acc, 0.f);
  }
  __syncthreads();
  const float* wt = weight + (long long)t * E;
  for (int c = threadIdx.x; c < C; c += 128) {
    float acc = xs[c];                      // residual sp
    for (int ee = 0; ee < E; ++ee) {
      float we = wt[ee];
      if (we != 0.f) {
        const float* w2r = w2 + ((long long)ee * C + c) * 16;
        const float* hr  = hs + ee * 16;
        float d = b2[(long long)ee * C + c];
        for (int j = 0; j < 16; ++j) d += hr[j] * w2r[j];
        acc += we * d;
      }
    }
    out[((long long)b * C + c) * P + p] = acc;
  }
}

// (lb1 + lb2)/2: lb = sum_e (usage_e/T)^2 * E
__global__ void k_lb_final(const float* __restrict__ u1, const float* __restrict__ u2,
                           float T, int E, float* __restrict__ outp)
{
  if (blockIdx.x == 0 && threadIdx.x == 0) {
    float l1 = 0.f, l2 = 0.f;
    for (int e = 0; e < E; ++e) {
      float a = u1[e] / T; l1 += a * a;
      float b = u2[e] / T; l2 += b * b;
    }
    outp[0] = 0.5f * (l1 * (float)E + l2 * (float)E);
  }
}

// ---------------------------------------------------------------------------
// Host side
// ---------------------------------------------------------------------------
namespace {

constexpr int Bn = 8, C = 256, Nn = 2048, Pp = 256, Ee = 8, HEADc = 8, HD = 32;
constexpr int Tt = Bn * Pp;

struct AttnP {
  const float *bn_b, *bn_g, *c1_b, *c1_w, *c2_b, *c2_w,
              *k_b, *k_w, *mh_b, *mh_w, *q_b, *q_w, *v_b, *v_w;
};
struct MoeP {
  const float *b1, *b2, *g1_b, *g1_w, *g2_b, *g2_w, *gbn_b, *gbn_g, *w1, *w2;
};

inline const float* F(void* const* d, int i) { return (const float*)d[i]; }

AttnP attn_alpha(void* const* d, int base) {
  AttnP p;
  p.bn_b = F(d, base + 0);  p.bn_g = F(d, base + 1);
  p.c1_b = F(d, base + 2);  p.c1_w = F(d, base + 3);
  p.c2_b = F(d, base + 4);  p.c2_w = F(d, base + 5);
  p.k_b  = F(d, base + 6);  p.k_w  = F(d, base + 7);
  p.mh_b = F(d, base + 8);  p.mh_w = F(d, base + 9);
  p.q_b  = F(d, base + 10); p.q_w  = F(d, base + 11);
  p.v_b  = F(d, base + 12); p.v_w  = F(d, base + 13);
  return p;
}
AttnP attn_ins(void* const* d, int base) {
  AttnP p;
  p.q_w  = F(d, base + 0);  p.q_b  = F(d, base + 1);
  p.k_w  = F(d, base + 2);  p.k_b  = F(d, base + 3);
  p.v_w  = F(d, base + 4);  p.v_b  = F(d, base + 5);
  p.mh_w = F(d, base + 6);  p.mh_b = F(d, base + 7);
  p.c1_w = F(d, base + 8);  p.c1_b = F(d, base + 9);
  p.bn_g = F(d, base + 10); p.bn_b = F(d, base + 11);
  p.c2_w = F(d, base + 12); p.c2_b = F(d, base + 13);
  return p;
}
MoeP moe_alpha(void* const* d, int base) {
  MoeP p;
  p.b1 = F(d, base + 0);   p.b2 = F(d, base + 1);
  p.g1_b = F(d, base + 2); p.g1_w = F(d, base + 3);
  p.g2_b = F(d, base + 4); p.g2_w = F(d, base + 5);
  p.gbn_b = F(d, base + 6); p.gbn_g = F(d, base + 7);
  p.w1 = F(d, base + 8);   p.w2 = F(d, base + 9);
  return p;
}
MoeP moe_ins(void* const* d, int base) {
  MoeP p;
  p.g1_w = F(d, base + 0); p.g1_b = F(d, base + 1);
  p.gbn_g = F(d, base + 2); p.gbn_b = F(d, base + 3);
  p.g2_w = F(d, base + 4); p.g2_b = F(d, base + 5);
  p.w1 = F(d, base + 6);   p.b1 = F(d, base + 7);
  p.w2 = F(d, base + 8);   p.b2 = F(d, base + 9);
  return p;
}

struct Net {
  const unsigned char* mask;
  const float* x;
  AttnP cluster, decl, ctx;
  MoeP moe1, moe2;
  const float *pool_bn_b, *pool_bn_g, *pool_c_b, *pool_c_w;
  const float *r_bn_b, *r_bn_g, *r_c1_b, *r_c1_w, *r_c2_b, *r_c2_w;
  const float *m1_b, *m1_w, *m2_b, *m2_w, *mbn_b, *mbn_g;
};

Net build_net(void* const* d, const int* sz) {
  Net n;
  if (sz[0] == Bn * Nn) {
    // jax tree order: alphabetical keys at every dict level.
    // top: mask, params{cluster,decl,moe1,moe2,pool,rect}, x
    n.mask = (const unsigned char*)d[0];
    n.cluster = attn_alpha(d, 1);
    n.decl    = attn_alpha(d, 15);
    n.moe1    = moe_alpha(d, 29);
    n.moe2    = moe_alpha(d, 39);
    n.pool_bn_b = F(d, 49); n.pool_bn_g = F(d, 50);
    n.pool_c_b  = F(d, 51); n.pool_c_w  = F(d, 52);
    n.r_bn_b = F(d, 53); n.r_bn_g = F(d, 54);
    n.r_c1_b = F(d, 55); n.r_c1_w = F(d, 56);
    n.r_c2_b = F(d, 57); n.r_c2_w = F(d, 58);
    n.ctx = attn_alpha(d, 59);
    n.m1_b = F(d, 73); n.m1_w = F(d, 74);
    n.m2_b = F(d, 75); n.m2_w = F(d, 76);
    n.mbn_b = F(d, 77); n.mbn_g = F(d, 78);
    n.x = F(d, 79);
  } else {
    // dict insertion order: x, mask, params{pool,moe1,cluster,rect,moe2,decl}
    n.x = F(d, 0);
    n.mask = (const unsigned char*)d[1];
    n.pool_bn_g = F(d, 2); n.pool_bn_b = F(d, 3);
    n.pool_c_w  = F(d, 4); n.pool_c_b  = F(d, 5);
    n.moe1    = moe_ins(d, 6);
    n.cluster = attn_ins(d, 16);
    n.ctx     = attn_ins(d, 30);
    n.m1_w = F(d, 44); n.m1_b = F(d, 45);
    n.mbn_g = F(d, 46); n.mbn_b = F(d, 47);
    n.m2_w = F(d, 48); n.m2_b = F(d, 49);
    n.r_c1_w = F(d, 50); n.r_c1_b = F(d, 51);
    n.r_bn_g = F(d, 52); n.r_bn_b = F(d, 53);
    n.r_c2_w = F(d, 54); n.r_c2_b = F(d, 55);
    n.moe2 = moe_ins(d, 56);
    n.decl = attn_ins(d, 66);
  }
  return n;
}

void gemm(hipStream_t st, int batch, int M, int N, int K,
          const float* A, long long sA, int lda, int tA,
          const float* Bm, long long sB, int ldb, int tB,
          float* Cm, long long sC, int ldc,
          const float* bias, const float* R, long long sR, int ldr, int relu)
{
  if (M >= 128 && N >= 128) {
    dim3 g((N + 127) / 128, (M + 127) / 128, batch);
    k_gemm<128, 128, 32, 64><<<g, 256, 0, st>>>(
        A, sA, lda, tA, Bm, sB, ldb, tB, Cm, sC, ldc,
        bias, R, sR, ldr, M, N, K, relu);
  } else {
    dim3 g((N + 63) / 64, (M + 63) / 64, batch);
    k_gemm<64, 64, 16, 32><<<g, 256, 0, st>>>(
        A, sA, lda, tA, Bm, sB, ldb, tB, Cm, sC, ldc,
        bias, R, sR, ldr, M, N, K, relu);
  }
}

struct Scratch {
  float *qc, *kc, *vc, *add, *cat, *h, *slab;
};

// attention(p, x1[N1], x2[N2], mask) -> out = x1 + c2(relu(bn(c1(cat(x1, mh)))))
void run_attention(hipStream_t st, const AttnP& p,
                   const float* x1, int N1, const float* x2, int N2,
                   const unsigned char* mask, float* out, const Scratch& s)
{
  gemm(st, Bn, C, N1, C, p.q_w, 0, C, 0, x1, (long long)C * N1, N1, 0,
       s.qc, (long long)C * N1, N1, p.q_b, nullptr, 0, 0, 0);
  gemm(st, Bn, C, N2, C, p.k_w, 0, C, 0, x2, (long long)C * N2, N2, 0,
       s.kc, (long long)C * N2, N2, p.k_b, nullptr, 0, 0, 0);
  gemm(st, Bn, C, N2, C, p.v_w, 0, C, 0, x2, (long long)C * N2, N2, 0,
       s.vc, (long long)C * N2, N2, p.v_b, nullptr, 0, 0, 0);

  const float scale = 0.17677669529663687f;   // 1/sqrt(hd=32)
  for (int g2 = 0; g2 < Bn * HEADc; ++g2) {
    const float* qg = s.qc + (long long)g2 * HD * N1;
    const float* kg = s.kc + (long long)g2 * HD * N2;
    const float* vg = s.vc + (long long)g2 * HD * N2;
    // qk[n,m] = sum_d q[d,n] k[d,m]   (transA, K = hd)
    gemm(st, 1, N1, N2, HD, qg, 0, N1, 1, kg, 0, N2, 0,
         s.slab, 0, N2, nullptr, nullptr, 0, 0, 0);
    const unsigned char* mrow = mask ? (mask + (long long)(g2 / HEADc) * N2) : nullptr;
    k_softmax<<<N1, 256, 0, st>>>(s.slab, s.slab, N2, scale, mrow);
    // add[d,n] = sum_m v[d,m] score[n,m]   (transB, K = N2)
    gemm(st, 1, HD, N1, N2, vg, 0, N2, 0, s.slab, 0, N2, 1,
         s.add + (long long)g2 * HD * N1, 0, N1, nullptr, nullptr, 0, 0, 0);
  }
  // mh projection (reuse qc as mh-out)
  gemm(st, Bn, C, N1, C, p.mh_w, 0, C, 0, s.add, (long long)C * N1, N1, 0,
       s.qc, (long long)C * N1, N1, p.mh_b, nullptr, 0, 0, 0);
  k_concat2<<<2048, 256, 0, st>>>(x1, s.qc, s.cat, Bn, C, N1);
  gemm(st, Bn, 2 * C, N1, 2 * C, p.c1_w, 0, 2 * C, 0, s.cat,
       (long long)2 * C * N1, N1, 0, s.h, (long long)2 * C * N1, N1,
       p.c1_b, nullptr, 0, 0, 0);
  k_bnorm<<<2 * C, 256, 0, st>>>(s.h, s.h, p.bn_g, p.bn_b, Bn, 2 * C, N1, 1, 1e-5f);
  gemm(st, Bn, C, N1, 2 * C, p.c2_w, 0, 2 * C, 0, s.h,
       (long long)2 * C * N1, N1, 0, out, (long long)C * N1, N1,
       p.c2_b, x1, (long long)C * N1, N1, 0);
}

void run_moe(hipStream_t st, const MoeP& p, float* sp,
             float* gbuf, float* logits, float* weight, float* usage)
{
  gemm(st, Bn, C, Pp, C, p.g1_w, 0, C, 0, sp, (long long)C * Pp, Pp, 0,
       gbuf, (long long)C * Pp, Pp, p.g1_b, nullptr, 0, 0, 0);
  k_bnorm<<<C, 256, 0, st>>>(gbuf, gbuf, p.gbn_g, p.gbn_b, Bn, C, Pp, 1, 1e-5f);
  gemm(st, Bn, Ee, Pp, C, p.g2_w, 0, C, 0, gbuf, (long long)C * Pp, Pp, 0,
       logits, (long long)Ee * Pp, Pp, p.g2_b, nullptr, 0, 0, 0);
  k_zero<<<1, 64, 0, st>>>(usage, Ee);
  k_moe_gate<<<(Tt + 255) / 256, 256, 0, st>>>(logits, weight, usage, Tt, Ee, Pp);
  k_moe_expert<<<Tt, 128, 0, st>>>(sp, weight, p.w1, p.b1, p.w2, p.b2,
                                   sp, Bn, C, Pp, Ee);
}

} // namespace

extern "C" void kernel_launch(void* const* d_in, const int* in_sizes, int n_in,
                              void* d_out, int out_size, void* d_ws, size_t ws_size,
                              hipStream_t stream)
{
  if (n_in < 66) return;
  Net net = build_net(d_in, in_sizes);

  const size_t CN  = (size_t)Bn * C * Nn;      // 4,194,304
  const size_t C2N = 2 * CN;                   // 8,388,608
  const size_t CP  = (size_t)Bn * C * Pp;      // 524,288

  float* f = (float*)d_ws;
  float* bufA = f;                 // CN
  float* bufB = bufA + CN;         // CN
  float* bufC = bufB + CN;         // CN
  float* bufD = bufC + CN;         // CN
  float* bufE = bufD + CN;         // 2CN
  float* bufF = bufE + C2N;        // 2CN
  float* bufG = bufF + C2N;        // CP (sp)
  float* bufH = bufG + CP;         // CP
  float* bufI = bufH + CP;         // CP
  float* slab = bufI + CP;         // CP (N1max*N2max = 2048*256)
  float* weight = slab + CP;       // T*E
  float* u1 = weight + (size_t)Tt * Ee;
  float* u2 = u1 + Ee;
  float* ybuf = u2 + Ee;           // B*C
  float* gate = ybuf + (size_t)Bn * C;   // B
  size_t need = (size_t)(gate + Bn - f) * sizeof(float);
  if (ws_size < need) return;

  float* out = (float*)d_out;
  Scratch s{bufA, bufB, bufC, bufD, bufE, bufF, slab};

  // ---- diff_pool: x3 -> sp (bufG) ----
  k_instnorm<<<Bn * C, 256, 0, stream>>>(net.x, bufA, Nn, 1e-3f);
  k_bnorm<<<C, 256, 0, stream>>>(bufA, bufA, net.pool_bn_g, net.pool_bn_b,
                                 Bn, C, Nn, 1, 1e-5f);
  gemm(stream, Bn, Pp, Nn, C, net.pool_c_w, 0, C, 0, bufA,
       (long long)C * Nn, Nn, 0, bufB, (long long)Pp * Nn, Nn,
       net.pool_c_b, nullptr, 0, 0, 0);
  k_softmax<<<Bn * Pp, 256, 0, stream>>>(bufB, bufB, Nn, 1.0f, nullptr);
  // sp[c,p] = sum_n x[c,n] * S[p,n]  (transB, K = N)
  gemm(stream, Bn, C, Pp, Nn, net.x, (long long)C * Nn, Nn, 0, bufB,
       (long long)Pp * Nn, Nn, 1, bufG, (long long)C * Pp, Pp,
       nullptr, nullptr, 0, 0, 0);

  // ---- moe1 (sp += moe(sp^T)) ----
  run_moe(stream, net.moe1, bufG, bufH, bufI, weight, u1);

  // ---- cluster attention: sp = attn(sp, x3, mask) -> bufH ----
  run_attention(stream, net.cluster, bufG, Pp, net.x, Nn, net.mask, bufH, s);

  // ---- rectify(bufH) -> bufG ----
  k_mean_rows<<<Bn * C, 256, 0, stream>>>(bufH, ybuf, Pp);
  k_rect_gate<<<1, 256, 0, stream>>>(ybuf, net.m1_w, net.m1_b,
                                     net.mbn_g, net.mbn_b,
                                     net.m2_w, net.m2_b, gate, Bn, C);
  k_scale_batch<<<1024, 256, 0, stream>>>(bufH, gate, bufI,
                                          (long long)C * Pp, (long long)Bn * C * Pp);
  run_attention(stream, net.ctx, bufH, Pp, bufH, Pp, nullptr, bufG, s); // out_context
  k_concat2<<<1024, 256, 0, stream>>>(bufI, bufG, bufE, Bn, C, Pp);
  gemm(stream, Bn, 2 * C, Pp, 2 * C, net.r_c1_w, 0, 2 * C, 0, bufE,
       (long long)2 * C * Pp, Pp, 0, bufF, (long long)2 * C * Pp, Pp,
       net.r_c1_b, nullptr, 0, 0, 0);
  k_bnorm<<<2 * C, 256, 0, stream>>>(bufF, bufF, net.r_bn_g, net.r_bn_b,
                                     Bn, 2 * C, Pp, 1, 1e-5f);
  gemm(stream, Bn, C, Pp, 2 * C, net.r_c2_w, 0, 2 * C, 0, bufF,
       (long long)2 * C * Pp, Pp, 0, bufG, (long long)C * Pp, Pp,
       net.r_c2_b, nullptr, 0, 0, 0);

  // ---- moe2 (sp += moe(sp^T)) ----
  run_moe(stream, net.moe2, bufG, bufH, bufI, weight, u2);

  // ---- decl attention: out = attn(x3, sp) -> d_out ----
  run_attention(stream, net.decl, net.x, Nn, bufG, Pp, nullptr, out, s);

  // ---- load-balance loss -> d_out[last] ----
  k_lb_final<<<1, 1, 0, stream>>>(u1, u2, (float)Tt, Ee, out + (out_size - 1));
}
